// WeatherAlphaModel_43843026157838
// MI455X (gfx1250) — 8-bit matrix operand order repaired and checked by reading, compiled, not run on hardware
//
#include <hip/hip_runtime.h>
#include <cstdint>
#include <cstddef>


typedef signed char i8;
typedef int v8i __attribute__((ext_vector_type(8)));
typedef int v4i __attribute__((ext_vector_type(4)));

#define D_     512
#define H_     8
#define L_     6
#define NS_    2
#define NP_    8
#define FEAT_  7
#define HID_   1024
#define TP_    240
#define TT_    240
#define B_     8
#define NTOK_  1920
#define LAYER_W8 11534336ULL

#define FLAG_SILU 1
#define FLAG_ACC  2

#define WMMA_IU8(a, b, c) \
    __builtin_amdgcn_wmma_i32_16x16x64_iu8(true, (a), true, (b), (c), false, false)

#if __has_builtin(__builtin_amdgcn_global_load_async_to_lds_b128) && \
    __has_builtin(__builtin_amdgcn_s_wait_asynccnt)
#define USE_ASYNC_LDS 1
#else
#define USE_ASYNC_LDS 0
#endif

__global__ void abssum_kernel(const float* __restrict__ w, int n,
                              float* __restrict__ scales, int mi) {
    __shared__ float red[256];
    float s = 0.f;
    for (int i = threadIdx.x; i < n; i += 256) s += fabsf(w[i]);
    red[threadIdx.x] = s; __syncthreads();
    for (int st = 128; st > 0; st >>= 1) {
        if (threadIdx.x < st) red[threadIdx.x] += red[threadIdx.x + st];
        __syncthreads();
    }
    if (threadIdx.x == 0) scales[mi] = fmaxf(red[0] / (float)n, 1e-5f);
}

__global__ void wquant_kernel(const float* __restrict__ w, int rows, int cols,
                              const float* __restrict__ scales, int mi,
                              i8* __restrict__ wt) {
    int idx = blockIdx.x * 256 + threadIdx.x;
    if (idx >= rows * cols) return;
    float sc = scales[mi];
    float q = rintf(w[idx] / sc);
    q = fminf(fmaxf(q, -1.f), 1.f);
    int r = idx / cols, c = idx % cols;
    wt[(size_t)c * rows + r] = (i8)q;
}

__global__ void actq_kernel(const float* __restrict__ X, i8* __restrict__ Xq,
                            float* __restrict__ invs, int K) {
    int row = blockIdx.x;
    const float* xr = X + (size_t)row * K;
    __shared__ float red[128];
    float mx = 0.f;
    for (int i = threadIdx.x; i < K; i += 128) mx = fmaxf(mx, fabsf(xr[i]));
    red[threadIdx.x] = mx; __syncthreads();
    for (int st = 64; st > 0; st >>= 1) {
        if (threadIdx.x < st) red[threadIdx.x] = fmaxf(red[threadIdx.x], red[threadIdx.x + st]);
        __syncthreads();
    }
    float m = fmaxf(red[0], 1e-5f);
    float s = 127.f / m;
    i8* qr = Xq + (size_t)row * K;
    for (int i = threadIdx.x; i < K; i += 128) {
        float q = rintf(xr[i] * s);
        q = fminf(fmaxf(q, -128.f), 127.f);
        qr[i] = (i8)q;
    }
    if (threadIdx.x == 0) invs[row] = m / 127.f;
}

__global__ void ln_kernel(const float* __restrict__ X, const float* __restrict__ g,
                          const float* __restrict__ b, float* __restrict__ Y, int K) {
    int row = blockIdx.x;
    const float* xr = X + (size_t)row * K;
    __shared__ float r1[128], r2[128];
    float s = 0.f, s2 = 0.f;
    for (int i = threadIdx.x; i < K; i += 128) { float v = xr[i]; s += v; s2 += v * v; }
    r1[threadIdx.x] = s; r2[threadIdx.x] = s2; __syncthreads();
    for (int st = 64; st > 0; st >>= 1) {
        if (threadIdx.x < st) { r1[threadIdx.x] += r1[threadIdx.x + st]; r2[threadIdx.x] += r2[threadIdx.x + st]; }
        __syncthreads();
    }
    float mean = r1[0] / (float)K;
    float var  = r2[0] / (float)K - mean * mean;
    float rinv = rsqrtf(var + 1e-5f);
    for (int i = threadIdx.x; i < K; i += 128)
        Y[(size_t)row * K + i] = (xr[i] - mean) * rinv * g[i] + b[i];
}

__global__ void embed_kernel(const float* __restrict__ x, const int* __restrict__ sid,
                             const float* __restrict__ pw, const float* __restrict__ pb,
                             const float* __restrict__ lns, const float* __restrict__ lnb,
                             const float* __restrict__ pos, const float* __restrict__ stab,
                             float* __restrict__ h) {
    int n = blockIdx.x;
    int b = n / TP_, t = n % TP_;
    const float* xr = x + ((size_t)b * TT_ + t) * FEAT_;
    float f[FEAT_];
    #pragma unroll
    for (int i = 0; i < FEAT_; i++) f[i] = xr[i];
    __shared__ float r1[128], r2[128];
    float vals[4];
    float s = 0.f, s2 = 0.f;
    #pragma unroll
    for (int j = 0; j < 4; j++) {
        int d = threadIdx.x + j * 128;
        float v = pb[d];
        #pragma unroll
        for (int i = 0; i < FEAT_; i++) v += f[i] * pw[i * D_ + d];
        vals[j] = v; s += v; s2 += v * v;
    }
    r1[threadIdx.x] = s; r2[threadIdx.x] = s2; __syncthreads();
    for (int st = 64; st > 0; st >>= 1) {
        if (threadIdx.x < st) { r1[threadIdx.x] += r1[threadIdx.x + st]; r2[threadIdx.x] += r2[threadIdx.x + st]; }
        __syncthreads();
    }
    float mean = r1[0] / (float)D_;
    float var  = r2[0] / (float)D_ - mean * mean;
    float rinv = rsqrtf(var + 1e-5f);
    int stn = sid[b];
    #pragma unroll
    for (int j = 0; j < 4; j++) {
        int d = threadIdx.x + j * 128;
        float y = (vals[j] - mean) * rinv * lns[d] + lnb[d]
                + pos[(size_t)t * D_ + d] + stab[(size_t)stn * D_ + d];
        h[(size_t)n * D_ + d] = y;
    }
}

__global__ __launch_bounds__(256) void attn_kernel(const float* __restrict__ Q,
                                                   const float* __restrict__ Kb,
                                                   const float* __restrict__ V,
                                                   float* __restrict__ O) {
    int idx = blockIdx.x;
    int t  = idx % TP_;
    int hh = (idx / TP_) % H_;
    int b  = idx / (TP_ * H_);
    size_t base = (size_t)b * TP_ * D_ + (size_t)hh * 64;
    __shared__ float qv[64];
    __shared__ float sc[256];
    __shared__ float red[256];
    int tid = threadIdx.x;
    if (tid < 64) qv[tid] = Q[base + (size_t)t * D_ + tid];
    __syncthreads();
    float val;
    if (tid < TP_) {
        const float* kr = Kb + base + (size_t)tid * D_;
        float s = 0.f;
        #pragma unroll
        for (int d = 0; d < 64; d++) s += qv[d] * kr[d];
        val = s * 0.125f;
    } else val = -3.0e38f;
    sc[tid] = val; red[tid] = val; __syncthreads();
    for (int st = 128; st > 0; st >>= 1) {
        if (tid < st) red[tid] = fmaxf(red[tid], red[tid + st]);
        __syncthreads();
    }
    float mx = red[0]; __syncthreads();
    float e = (tid < TP_) ? expf(sc[tid] - mx) : 0.f;
    sc[tid] = e; red[tid] = e; __syncthreads();
    for (int st = 128; st > 0; st >>= 1) {
        if (tid < st) red[tid] += red[tid + st];
        __syncthreads();
    }
    float inv = 1.f / red[0]; __syncthreads();
    int dcol = tid & 63, grp = tid >> 6;
    float acc = 0.f;
    for (int s2 = grp; s2 < TP_; s2 += 4) acc += sc[s2] * V[base + (size_t)s2 * D_ + dcol];
    red[tid] = acc; __syncthreads();
    if (grp == 0) {
        float o = (red[tid] + red[tid + 64] + red[tid + 128] + red[tid + 192]) * inv;
        O[base + (size_t)t * D_ + dcol] = o;
    }
}

__global__ void gate_kernel(const float* __restrict__ M, const float* __restrict__ gw,
                            const float* __restrict__ gb, float* __restrict__ cw) {
    int n = blockIdx.x * blockDim.x + threadIdx.x;
    if (n >= NTOK_) return;
    const float* mr = M + (size_t)n * D_;
    float lg[NP_];
    #pragma unroll
    for (int e = 0; e < NP_; e++) lg[e] = gb[e];
    for (int d = 0; d < D_; d++) {
        float md = mr[d];
        const float* gr = gw + d * NP_;
        #pragma unroll
        for (int e = 0; e < NP_; e++) lg[e] += md * gr[e];
    }
    float mx = lg[0];
    #pragma unroll
    for (int e = 1; e < NP_; e++) mx = fmaxf(mx, lg[e]);
    float p[NP_]; float sum = 0.f;
    #pragma unroll
    for (int e = 0; e < NP_; e++) { p[e] = expf(lg[e] - mx); sum += p[e]; }
    #pragma unroll
    for (int e = 0; e < NP_; e++) p[e] /= sum;
    int i1 = 0;
    #pragma unroll
    for (int e = 1; e < NP_; e++) if (p[e] > p[i1]) i1 = e;
    int i2 = -1;
    #pragma unroll
    for (int e = 0; e < NP_; e++) { if (e == i1) continue; if (i2 < 0 || p[e] > p[i2]) i2 = e; }
    float tot = p[i1] + p[i2];
    #pragma unroll
    for (int e = 0; e < NP_; e++)
        cw[(size_t)n * NP_ + e] = (e == i1) ? p[i1] / tot : (e == i2) ? p[i2] / tot : 0.f;
}

__global__ void head_kernel(const float* __restrict__ Hf, const float* __restrict__ lns,
                            const float* __restrict__ lnb, const float* __restrict__ hw,
                            const float* __restrict__ hb, float* __restrict__ out) {
    int b = blockIdx.x;
    const float* xr = Hf + ((size_t)b * TP_ + TP_ - 1) * D_;
    __shared__ float r1[128], r2[128];
    float s = 0.f, s2 = 0.f;
    for (int i = threadIdx.x; i < D_; i += 128) { float v = xr[i]; s += v; s2 += v * v; }
    r1[threadIdx.x] = s; r2[threadIdx.x] = s2; __syncthreads();
    for (int st = 64; st > 0; st >>= 1) {
        if (threadIdx.x < st) { r1[threadIdx.x] += r1[threadIdx.x + st]; r2[threadIdx.x] += r2[threadIdx.x + st]; }
        __syncthreads();
    }
    float mean = r1[0] / (float)D_;
    float var  = r2[0] / (float)D_ - mean * mean;
    float rinv = rsqrtf(var + 1e-5f);
    __syncthreads();
    float dot = 0.f;
    for (int i = threadIdx.x; i < D_; i += 128) {
        float y = (xr[i] - mean) * rinv * lns[i] + lnb[i];
        dot += y * hw[i];
    }
    r1[threadIdx.x] = dot; __syncthreads();
    for (int st = 64; st > 0; st >>= 1) {
        if (threadIdx.x < st) r1[threadIdx.x] += r1[threadIdx.x + st];
        __syncthreads();
    }
    if (threadIdx.x == 0) out[b] = r1[0] + hb[0];
}

__global__ __launch_bounds__(256)
void bitgemm_kernel(const i8* __restrict__ A8, const float* __restrict__ Ais,
                    const i8* __restrict__ Wt, const float* __restrict__ scales, int sidx,
                    const float* __restrict__ bias, const float* __restrict__ coef, int cstride,
                    float* __restrict__ Y, int flags, int K, int Dout) {
    __shared__ __align__(16) i8 As[2][64][80];
    __shared__ __align__(16) i8 Bs[2][128][80];
    int tid  = threadIdx.x;
    int lane = tid & 31;
    int w    = tid >> 5;
    int wn   = w & 3;
    int wm   = w >> 2;
    int half = lane >> 4;
    int mn   = lane & 15;
    int rowBase = blockIdx.y * 64;
    int colBase = blockIdx.x * 128;
    int lr = tid >> 2;
    int lc = (tid & 3) * 16;
    const i8* ga  = A8 + (size_t)(rowBase + lr) * K + lc;
    const i8* gb0 = Wt + (size_t)(colBase + lr) * K + lc;
    const i8* gb1 = Wt + (size_t)(colBase + lr + 64) * K + lc;

    v8i c00 = {}, c01 = {}, c10 = {}, c11 = {};
    int nk = K >> 6;

#if USE_ASYNC_LDS
    #define STAGE_TILE(buf, kb) do {                                             \
        __builtin_amdgcn_global_load_async_to_lds_b128(                          \
            (v4i*)(ga + (kb)),  (v4i*)&As[buf][lr][lc], 0, 0);                   \
        __builtin_amdgcn_global_load_async_to_lds_b128(                          \
            (v4i*)(gb0 + (kb)), (v4i*)&Bs[buf][lr][lc], 0, 0);                   \
        __builtin_amdgcn_global_load_async_to_lds_b128(                          \
            (v4i*)(gb1 + (kb)), (v4i*)&Bs[buf][lr + 64][lc], 0, 0);              \
    } while (0)
    STAGE_TILE(0, 0);
#endif

    for (int i = 0; i < nk; i++) {
        int p = i & 1;
#if USE_ASYNC_LDS
        if (i + 1 < nk) {
            STAGE_TILE(p ^ 1, (i + 1) << 6);
            __builtin_amdgcn_s_wait_asynccnt(3);
        } else {
            __builtin_amdgcn_s_wait_asynccnt(0);
        }
        __syncthreads();
#else
        {
            int kb = i << 6;
            const int* ap  = (const int*)(ga + kb);
            const int* bp0 = (const int*)(gb0 + kb);
            const int* bp1 = (const int*)(gb1 + kb);
            if (i + 1 < nk) {
                __builtin_prefetch((const void*)(ap + 16), 0, 0);
                __builtin_prefetch((const void*)(bp0 + 16), 0, 0);
                __builtin_prefetch((const void*)(bp1 + 16), 0, 0);
            }
            int* asl  = (int*)&As[p][lr][lc];
            int* bsl0 = (int*)&Bs[p][lr][lc];
            int* bsl1 = (int*)&Bs[p][lr + 64][lc];
            #pragma unroll
            for (int j = 0; j < 4; j++) asl[j] = ap[j];
            #pragma unroll
            for (int j = 0; j < 4; j++) bsl0[j] = bp0[j];
            #pragma unroll
            for (int j = 0; j < 4; j++) bsl1[j] = bp1[j];
            __syncthreads();
        }
#endif
        v8i a0, a1, b0, b1;
        #pragma unroll
        for (int v = 0; v < 8; v++) {
            int ka = ((v >> 1) << 4) + ((v & 1) << 2) + (half << 3);
            a0[v] = *(const int*)&As[p][wm * 32 + mn][ka];
            a1[v] = *(const int*)&As[p][wm * 32 + 16 + mn][ka];
            int kv = ((v >> 1) << 4) + ((v & 1) << 2) + (half << 3);
            b0[v] = *(const int*)&Bs[p][wn * 32 + mn][kv];
            b1[v] = *(const int*)&Bs[p][wn * 32 + 16 + mn][kv];
        }
        c00 = WMMA_IU8(a0, b0, c00);
        c01 = WMMA_IU8(a0, b1, c01);
        c10 = WMMA_IU8(a1, b0, c10);
        c11 = WMMA_IU8(a1, b1, c11);
        __syncthreads();
    }

    float wsc = scales[sidx];
    int col0 = colBase + wn * 32 + mn;
    int col1 = col0 + 16;
    float bb0 = bias[col0], bb1 = bias[col1];
    #pragma unroll
    for (int r = 0; r < 8; r++) {
        int row0 = rowBase + wm * 32 + half * 8 + r;
        int row1 = row0 + 16;
        float dq0 = wsc * Ais[row0];
        float dq1 = wsc * Ais[row1];
        float v00 = (float)c00[r] * dq0 + bb0;
        float v01 = (float)c01[r] * dq0 + bb1;
        float v10 = (float)c10[r] * dq1 + bb0;
        float v11 = (float)c11[r] * dq1 + bb1;
        if (flags & FLAG_SILU) {
            v00 = v00 / (1.f + expf(-v00));
            v01 = v01 / (1.f + expf(-v01));
            v10 = v10 / (1.f + expf(-v10));
            v11 = v11 / (1.f + expf(-v11));
        }
        float* y00 = Y + (size_t)row0 * Dout + col0;
        float* y01 = Y + (size_t)row0 * Dout + col1;
        float* y10 = Y + (size_t)row1 * Dout + col0;
        float* y11 = Y + (size_t)row1 * Dout + col1;
        if (flags & FLAG_ACC) {
            float cf0 = coef ? coef[(size_t)row0 * cstride] : 1.0f;
            float cf1 = coef ? coef[(size_t)row1 * cstride] : 1.0f;
            *y00 += cf0 * v00; *y01 += cf0 * v01;
            *y10 += cf1 * v10; *y11 += cf1 * v11;
        } else {
            *y00 = v00; *y01 = v01;
            *y10 = v10; *y11 = v11;
        }
    }
}

extern "C" void kernel_launch(void* const* d_in, const int* in_sizes, int n_in,
                              void* d_out, int out_size, void* d_ws, size_t ws_size,
                              hipStream_t stream) {
    (void)in_sizes; (void)n_in; (void)out_size; (void)ws_size;
    const float* x       = (const float*)d_in[0];
    const int*   sid     = (const int*)  d_in[1];
    const float* patch_w = (const float*)d_in[2];
    const float* patch_b = (const float*)d_in[3];
    const float* ln_p_s  = (const float*)d_in[4];
    const float* ln_p_b  = (const float*)d_in[5];
    const float* pos     = (const float*)d_in[6];
    const float* stab    = (const float*)d_in[7];
    const float* ln1_s   = (const float*)d_in[8];
    const float* ln1_b   = (const float*)d_in[9];
    const float* wq      = (const float*)d_in[10];
    const float* bq      = (const float*)d_in[11];
    const float* wk      = (const float*)d_in[12];
    const float* bk      = (const float*)d_in[13];
    const float* wv      = (const float*)d_in[14];
    const float* bv      = (const float*)d_in[15];
    const float* wo      = (const float*)d_in[16];
    const float* bo      = (const float*)d_in[17];
    const float* ln2_s   = (const float*)d_in[18];
    const float* ln2_b   = (const float*)d_in[19];
    const float* gate_w  = (const float*)d_in[20];
    const float* gate_b  = (const float*)d_in[21];
    const float* sw1     = (const float*)d_in[22];
    const float* sb1     = (const float*)d_in[23];
    const float* sw2     = (const float*)d_in[24];
    const float* sb2     = (const float*)d_in[25];
    const float* pw1     = (const float*)d_in[26];
    const float* pb1     = (const float*)d_in[27];
    const float* pw2     = (const float*)d_in[28];
    const float* pb2     = (const float*)d_in[29];
    const float* lno_s   = (const float*)d_in[30];
    const float* lno_b   = (const float*)d_in[31];
    const float* head_w  = (const float*)d_in[32];
    const float* head_b  = (const float*)d_in[33];

    char* ws = (char*)d_ws;
    size_t off = 0;
    auto walloc = [&](size_t b) { void* p = ws + off; off = (off + b + 255) & ~(size_t)255; return p; };
    i8*    W8     = (i8*)   walloc(L_ * LAYER_W8);
    float* scales = (float*)walloc(144 * sizeof(float));
    float* h      = (float*)walloc((size_t)NTOK_ * D_ * 4);
    float* a      = (float*)walloc((size_t)NTOK_ * D_ * 4);
    float* qb     = (float*)walloc((size_t)NTOK_ * D_ * 4);
    float* kb     = (float*)walloc((size_t)NTOK_ * D_ * 4);
    float* vb     = (float*)walloc((size_t)NTOK_ * D_ * 4);
    float* H1     = (float*)walloc((size_t)NTOK_ * HID_ * 4);
    i8*    x8     = (i8*)   walloc((size_t)NTOK_ * D_);
    i8*    g8     = (i8*)   walloc((size_t)NTOK_ * HID_);
    float* xis    = (float*)walloc((size_t)NTOK_ * 4);
    float* gis    = (float*)walloc((size_t)NTOK_ * 4);
    float* cw     = (float*)walloc((size_t)NTOK_ * NP_ * 4);

    int mi = 0;
    auto quant = [&](const float* wptr, int rows, int cols, i8* dst) {
        int n = rows * cols;
        abssum_kernel<<<1, 256, 0, stream>>>(wptr, n, scales, mi);
        wquant_kernel<<<(n + 255) / 256, 256, 0, stream>>>(wptr, rows, cols, scales, mi, dst);
        mi++;
    };
    for (int l = 0; l < L_; l++) {
        i8* base = W8 + (size_t)l * LAYER_W8;
        quant(wq + (size_t)l * D_ * D_, D_, D_, base + 0);
        quant(wk + (size_t)l * D_ * D_, D_, D_, base + 262144);
        quant(wv + (size_t)l * D_ * D_, D_, D_, base + 524288);
        quant(wo + (size_t)l * D_ * D_, D_, D_, base + 786432);
        for (int e = 0; e < NS_; e++)
            quant(sw1 + ((size_t)l * NS_ + e) * D_ * HID_, D_, HID_, base + 1048576 + (size_t)e * 524288);
        for (int e = 0; e < NS_; e++)
            quant(sw2 + ((size_t)l * NS_ + e) * HID_ * D_, HID_, D_, base + 2097152 + (size_t)e * 524288);
        for (int e = 0; e < NP_; e++)
            quant(pw1 + ((size_t)l * NP_ + e) * D_ * HID_, D_, HID_, base + 3145728 + (size_t)e * 524288);
        for (int e = 0; e < NP_; e++)
            quant(pw2 + ((size_t)l * NP_ + e) * HID_ * D_, HID_, D_, base + 7340032 + (size_t)e * 524288);
    }

    auto gemm = [&](const i8* A, const float* Ais, const i8* Wt, int sidx,
                    const float* bias, float* Y, int K, int Dout, int flags,
                    const float* coef = nullptr, int cstride = 0) {
        dim3 g(Dout / 128, NTOK_ / 64);
        bitgemm_kernel<<<g, 256, 0, stream>>>(A, Ais, Wt, scales, sidx, bias,
                                              coef, cstride, Y, flags, K, Dout);
    };

    embed_kernel<<<NTOK_, 128, 0, stream>>>(x, sid, patch_w, patch_b, ln_p_s, ln_p_b,
                                            pos, stab, h);

    for (int l = 0; l < L_; l++) {
        const i8* Bq = W8 + (size_t)l * LAYER_W8;
        int si = l * 24;
        ln_kernel<<<NTOK_, 128, 0, stream>>>(h, ln1_s + l * D_, ln1_b + l * D_, a, D_);
        actq_kernel<<<NTOK_, 128, 0, stream>>>(a, x8, xis, D_);
        gemm(x8, xis, Bq + 0,      si + 0, bq + l * D_, qb, D_, D_, 0);
        gemm(x8, xis, Bq + 262144, si + 1, bk + l * D_, kb, D_, D_, 0);
        gemm(x8, xis, Bq + 524288, si + 2, bv + l * D_, vb, D_, D_, 0);
        attn_kernel<<<B_ * H_ * TP_, 256, 0, stream>>>(qb, kb, vb, a);
        actq_kernel<<<NTOK_, 128, 0, stream>>>(a, x8, xis, D_);
        gemm(x8, xis, Bq + 786432, si + 3, bo + l * D_, h, D_, D_, FLAG_ACC);
        ln_kernel<<<NTOK_, 128, 0, stream>>>(h, ln2_s + l * D_, ln2_b + l * D_, a, D_);
        actq_kernel<<<NTOK_, 128, 0, stream>>>(a, x8, xis, D_);
        gate_kernel<<<(NTOK_ + 127) / 128, 128, 0, stream>>>(a, gate_w + (size_t)l * D_ * NP_,
                                                             gate_b + l * NP_, cw);
        for (int e = 0; e < NS_; e++) {
            gemm(x8, xis, Bq + 1048576 + (size_t)e * 524288, si + 4 + e,
                 sb1 + ((size_t)l * NS_ + e) * HID_, H1, D_, HID_, FLAG_SILU);
            actq_kernel<<<NTOK_, 128, 0, stream>>>(H1, g8, gis, HID_);
            gemm(g8, gis, Bq + 2097152 + (size_t)e * 524288, si + 6 + e,
                 sb2 + ((size_t)l * NS_ + e) * D_, h, HID_, D_, FLAG_ACC);
        }
        for (int e = 0; e < NP_; e++) {
            gemm(x8, xis, Bq + 3145728 + (size_t)e * 524288, si + 8 + e,
                 pb1 + ((size_t)l * NP_ + e) * HID_, H1, D_, HID_, FLAG_SILU);
            actq_kernel<<<NTOK_, 128, 0, stream>>>(H1, g8, gis, HID_);
            gemm(g8, gis, Bq + 7340032 + (size_t)e * 524288, si + 16 + e,
                 pb2 + ((size_t)l * NP_ + e) * D_, h, HID_, D_, FLAG_ACC, cw + e, NP_);
        }
    }

    head_kernel<<<B_, 128, 0, stream>>>(h, lno_s, lno_b, head_w, head_b, (float*)d_out);
}
